// GNNScheduler_54700703482224
// MI455X (gfx1250) — compile-verified, matrix operand order checked by reading, not run on hardware
//
#include <hip/hip_runtime.h>
#include <cmath>
#include <cstdint>

typedef __attribute__((ext_vector_type(16))) _Float16 v16h;
typedef __attribute__((ext_vector_type(8)))  _Float16 v8h;
typedef __attribute__((ext_vector_type(4)))  _Float16 v4h;
typedef __attribute__((ext_vector_type(8)))  float    v8f;
typedef __attribute__((ext_vector_type(4)))  float    v4f;

static constexpr int NNODES = 30000;
static constexpr int NEDGES = 480000;

#define DEV_INLINE __device__ __forceinline__

// ---------------------------------------------------------------- utilities
__global__ void fill_kernel(float* __restrict__ p, float v, long long n) {
  long long i = (long long)blockIdx.x * blockDim.x + threadIdx.x;
  if (i < n) p[i] = v;
}

// in: [T][K][N] f32  ->  out: [T][N][K] f16  (one-time weight prep)
__global__ void convert_tr_kernel(const float* __restrict__ in,
                                  _Float16* __restrict__ out,
                                  int K, int N, int T) {
  long long idx = (long long)blockIdx.x * blockDim.x + threadIdx.x;
  long long total = (long long)T * K * N;
  if (idx >= total) return;
  int n = (int)(idx % N);
  int k = (int)((idx / N) % K);
  int t = (int)(idx / ((long long)N * K));
  out[((size_t)t * N + n) * K + k] = (_Float16)in[((size_t)t * K + k) * N + n];
}

DEV_INLINE void atomicMaxF(float* addr, float val) {
  unsigned int* ia = (unsigned int*)addr;
  unsigned int cur = __float_as_uint(*addr);
  while (__uint_as_float(cur) < val) {
    unsigned int prev = atomicCAS(ia, cur, __float_as_uint(val));
    if (prev == cur) break;
    cur = prev;
  }
}

// ---------------------------------------------------------------- WMMA GEMM
// C[M,N] = act( A[M,K](opt relu) @ B[K,N] );  A f32 row-major, B pre-converted
// f16 transposed [N][K].  Block: 256 thr = 8 wave32, tile 64(M) x 64(N), BK=32,
// double-buffered LDS, each wave owns 16x32 (2 WMMA / K-step, 16 WMMA / block).
#define BM 64
#define BN 64
#define BK 32
#define LDK (BK + 8)   // 40 halves = 80B row stride (16B aligned)

__global__ __launch_bounds__(256) void gemm_wmma_kernel(
    const float* __restrict__ A, const _Float16* __restrict__ Bt,
    float* __restrict__ C, int M, int K,
    int lda, int ldc, int relu_a, int relu_out) {
  __shared__ alignas(16) _Float16 sA[2][BM][LDK];
  __shared__ alignas(16) _Float16 sB[2][BN][LDK];

  const int tid  = threadIdx.x;
  const int wave = tid >> 5;
  const int lane = tid & 31;
  const int bm = blockIdx.x * BM;
  const int bn = blockIdx.y * BN;
  const int wm  = (wave >> 1) * 16;   // 4 M-tiles
  const int wn0 = (wave & 1) * 32;    // 2 N-cols x 2 tiles each

  v8f acc0 = {0.f,0.f,0.f,0.f,0.f,0.f,0.f,0.f};
  v8f acc1 = {0.f,0.f,0.f,0.f,0.f,0.f,0.f,0.f};
  const int mrow = lane & 15;
  const int kb   = (lane < 16) ? 0 : 8;   // ISA wave32 16-bit fragment striping

  // staging lambdas: no bounds checks except M-row clamp (K,N tile-exact)
  auto stage = [&](int buf, int k0) {
    // A: 64x32 f32 -> f16, 512 float4 slots, 2 per thread
#pragma unroll
    for (int s = tid; s < (BM * BK) / 4; s += 256) {
      int r  = s >> 3;
      int c4 = (s & 7) << 2;
      int gm = bm + r; gm = gm < M ? gm : (M - 1);
      const float* ap = A + (size_t)gm * lda + k0 + c4;
      v4f v = *(const v4f*)ap;
      __builtin_prefetch(ap + BK, 0, 1);
      if (relu_a) {
#pragma unroll
        for (int j = 0; j < 4; j++) v[j] = v[j] > 0.f ? v[j] : 0.f;
      }
      v4h hv;
#pragma unroll
      for (int j = 0; j < 4; j++) hv[j] = (_Float16)v[j];
      *(v4h*)&sA[buf][r][c4] = hv;
    }
    // B: 64 rows x 32 halves, pure v8h copies, 256 slots, 1 per thread
#pragma unroll
    for (int s = tid; s < (BN * BK) / 8; s += 256) {
      int n = s >> 2;
      int g = (s & 3) << 3;
      *(v8h*)&sB[buf][n][g] = *(const v8h*)&Bt[(size_t)(bn + n) * K + k0 + g];
    }
  };

  const int nk = K >> 5;
  stage(0, 0);
  __syncthreads();

  for (int kt = 0; kt < nk; kt++) {
    int cur = kt & 1;
    if (kt + 1 < nk) stage(cur ^ 1, (kt + 1) * BK);

    union { v16h v; v8h h[2]; } fa, fb0, fb1;
    fa.h[0]  = *(const v8h*)&sA[cur][wm + mrow][kb];
    fa.h[1]  = *(const v8h*)&sA[cur][wm + mrow][kb + 16];
    fb0.h[0] = *(const v8h*)&sB[cur][wn0 + mrow][kb];
    fb0.h[1] = *(const v8h*)&sB[cur][wn0 + mrow][kb + 16];
    fb1.h[0] = *(const v8h*)&sB[cur][wn0 + 16 + mrow][kb];
    fb1.h[1] = *(const v8h*)&sB[cur][wn0 + 16 + mrow][kb + 16];
    acc0 = __builtin_amdgcn_wmma_f32_16x16x32_f16(
        false, fa.v, false, fb0.v, (short)0, acc0, false, false);
    acc1 = __builtin_amdgcn_wmma_f32_16x16x32_f16(
        false, fa.v, false, fb1.v, (short)0, acc1, false, false);
    __syncthreads();
  }

  // epilogue: lane = column, VGPR r = row r + 8*(lane>=16)
  int col0  = bn + wn0 + (lane & 15);
  int mbase = bm + wm + ((lane >> 4) << 3);
#pragma unroll
  for (int r = 0; r < 8; r++) {
    int m = mbase + r;
    if (m < M) {
      float v0 = acc0[r], v1 = acc1[r];
      if (relu_out) { v0 = v0 > 0.f ? v0 : 0.f; v1 = v1 > 0.f ? v1 : 0.f; }
      C[(size_t)m * ldc + col0]      = v0;
      C[(size_t)m * ldc + col0 + 16] = v1;
    }
  }
}

// ---------------------------------------------------------------- GAT pieces
// one wave per (node, edge-type, head): coalesced lane-strided dot + shuffle
__global__ void sisj_kernel(const float* __restrict__ lin, const float* __restrict__ ew,
                            float* __restrict__ si, float* __restrict__ sj,
                            int N, int H, int outc) {
  int wid  = blockIdx.x * (blockDim.x >> 5) + (threadIdx.x >> 5);
  int lane = threadIdx.x & 31;
  int total = N * 5 * H;
  if (wid >= total) return;
  int h = wid % H;
  int t = (wid / H) % 5;
  int n = wid / (5 * H);
  const float* xr = lin + ((size_t)n * H + h) * outc;
  const float* wi = ew + ((size_t)t * H + h) * 2 * outc;
  const float* wj = wi + outc;
  float a = 0.f, b = 0.f;
  for (int c = lane; c < outc; c += 32) {
    float v = xr[c];
    a += v * wi[c];
    b += v * wj[c];
  }
#pragma unroll
  for (int m = 16; m; m >>= 1) { a += __shfl_xor(a, m); b += __shfl_xor(b, m); }
  if (lane == 0) { si[wid] = a; sj[wid] = b; }
}

__global__ void edge_pass1_kernel(const int* __restrict__ ei, const int* __restrict__ eti,
                                  const float* __restrict__ si, const float* __restrict__ sj,
                                  float* __restrict__ alpha, float* __restrict__ amax,
                                  int E, int h_sh) {
  int idx = blockIdx.x * blockDim.x + threadIdx.x;
  if (idx >= (E << h_sh)) return;
  int e = idx >> h_sh, h = idx & ((1 << h_sh) - 1);
  int H = 1 << h_sh;
  int src = ei[e], dst = ei[E + e];
  int t = (e >= eti[1]) + (e >= eti[2]) + (e >= eti[3]) + (e >= eti[4]);
  float a = si[((size_t)dst * 5 + t) * H + h] + sj[((size_t)src * 5 + t) * H + h];
  a = a >= 0.f ? a : 0.2f * a;              // leaky relu
  alpha[idx] = a;
  atomicMaxF(&amax[(dst << h_sh) + h], a);
}

__global__ void edge_pass2_kernel(const int* __restrict__ ei, const float* __restrict__ alpha,
                                  const float* __restrict__ amax, float* __restrict__ exv,
                                  float* __restrict__ den, int E, int h_sh) {
  int idx = blockIdx.x * blockDim.x + threadIdx.x;
  if (idx >= (E << h_sh)) return;
  int e = idx >> h_sh, h = idx & ((1 << h_sh) - 1);
  int dst = ei[E + e];
  float ev = __expf(alpha[idx] - amax[(dst << h_sh) + h]);
  exv[idx] = ev;
  atomicAdd(&den[(dst << h_sh) + h], ev);
}

__global__ void edge_pass3_kernel(const int* __restrict__ ei, const float* __restrict__ exv,
                                  const float* __restrict__ den, const float* __restrict__ lin,
                                  float* __restrict__ aggr, int E, int h_sh, int oc_sh) {
  long long idx = (long long)blockIdx.x * blockDim.x + threadIdx.x;
  long long total = (long long)E << (h_sh + oc_sh);
  if (idx >= total) return;
  int c = (int)(idx & ((1 << oc_sh) - 1));
  int h = (int)((idx >> oc_sh) & ((1 << h_sh) - 1));
  int e = (int)(idx >> (oc_sh + h_sh));
  int src = ei[e], dst = ei[E + e];
  float att = exv[((size_t)e << h_sh) + h] / (den[(dst << h_sh) + h] + 1e-16f);
  atomicAdd(&aggr[((((size_t)dst << h_sh) + h) << oc_sh) + c],
            lin[((((size_t)src << h_sh) + h) << oc_sh) + c] * att);
}

// out = (relu?) (aggr + nb[type(n)] (+ resid))
__global__ void bias_out_kernel(const float* __restrict__ aggr, const float* __restrict__ nb,
                                const int* __restrict__ nti, const float* __restrict__ resid,
                                float* __restrict__ out, int N, int hc, int do_relu) {
  long long idx = (long long)blockIdx.x * blockDim.x + threadIdx.x;
  if (idx >= (long long)N * hc) return;
  int n = (int)(idx / hc), j = (int)(idx % hc);
  int t = (n >= nti[1]) + (n >= nti[2]) + (n >= nti[3]);
  float v = aggr[idx] + nb[(size_t)t * hc + j];
  if (resid) v += resid[idx];
  if (do_relu) v = v > 0.f ? v : 0.f;
  out[idx] = v;
}

// column means of a[r0:r1, 0:64]
__global__ void col_mean_kernel(const float* __restrict__ a, int r0, int r1, int ld,
                                float* __restrict__ out) {
  __shared__ float red[256];
  int f = blockIdx.x;
  float s = 0.f;
  for (int r = r0 + threadIdx.x; r < r1; r += 256) s += a[(size_t)r * ld + f];
  red[threadIdx.x] = s;
  __syncthreads();
  for (int st = 128; st; st >>= 1) {
    if ((int)threadIdx.x < st) red[threadIdx.x] += red[threadIdx.x + st];
    __syncthreads();
  }
  if (threadIdx.x == 0) out[f] = red[0] / (float)(r1 - r0);
}

// ---------------------------------------------------------------- fused MLP head
// One block per node: 32 choice rows, full MLP + softmax; h = relu(t@w1+b1) via
// WMMA with double-buffered pre-transposed f16 w1 chunks.
union alignas(16) WHU {
  _Float16 w[2][256][32];   // staged w1 chunks [n][k] (double buffered)
  float    h[32][256];      // hidden activations (reused after all WMMA done)
};

__global__ __launch_bounds__(256) void mlp_head_kernel(
    const float* __restrict__ choices,   // [nodes][32][lc]
    const float* __restrict__ act,       // [.., 64] stride act_stride (0 = bcast)
    int act_stride, int lc,
    const float* __restrict__ prm,       // mlp block base (b1,b2,lb,lw,rb,rw,w1,w2)
    const _Float16* __restrict__ w1t,    // pre-converted f16 w1 transposed [N=256][K=256]
    float* __restrict__ out) {           // [nodes][32]
  const float* b1 = prm;
  const float* b2 = prm + 256;
  const float* lb = prm + 257;
  const float* lw = prm + 385;
  const float* rb = lw + lc * 128;
  const float* rw = rb + 128;
  const float* w2 = rw + 64 * 128 + 256 * 256;   // skip f32 w1

  __shared__ alignas(16) _Float16 sT[32][256];
  __shared__ WHU u;
  __shared__ float sAct[64];
  __shared__ float sR[128];
  __shared__ float sLogit[32];

  const int tid  = threadIdx.x;
  const int node = blockIdx.x;
  const int wave = tid >> 5;
  const int lane = tid & 31;

  if (tid < 64) sAct[tid] = act[(size_t)node * act_stride + tid];
  __syncthreads();

  // r-branch (shared by all 32 choices of this node)
  if (tid < 128) {
    float s = rb[tid];
    for (int q = 0; q < 64; q++) s += sAct[q] * rw[q * 128 + tid];
    sR[tid] = s > 0.f ? s : 0.f;
  }
  __syncthreads();

  // l-branch + pack t = relu([l, r]) as f16
  for (int i = tid; i < 32 * 128; i += 256) {
    int c = i >> 7, f = i & 127;
    float s = lb[f];
    const float* ch = choices + ((size_t)node * 32 + c) * lc;
    for (int q = 0; q < lc; q++) s += ch[q] * lw[q * 128 + f];
    sT[c][f]       = (_Float16)(s > 0.f ? s : 0.f);
    sT[c][128 + f] = (_Float16)sR[f];
  }

  auto stageW = [&](int buf, int k0) {
#pragma unroll
    for (int s = tid; s < 1024; s += 256) {           // 8192 halves / 8
      int n = s >> 2, g = (s & 3) << 3;
      *(v8h*)&u.w[buf][n][g] = *(const v8h*)&w1t[(size_t)n * 256 + k0 + g];
    }
  };

  stageW(0, 0);
  __syncthreads();

  // h = relu(t @ w1 + b1): M=32, N=256, K=256 via WMMA f16
  const int m_tile = wave & 1;     // 2 M-tiles
  const int n_col  = wave >> 1;    // 4 wave-columns over 16 N-tiles
  v8f acc[4];
#pragma unroll
  for (int j = 0; j < 4; j++) acc[j] = (v8f){0.f,0.f,0.f,0.f,0.f,0.f,0.f,0.f};
  const int mrow = lane & 15;
  const int kb   = (lane < 16) ? 0 : 8;

  for (int kt = 0; kt < 8; kt++) {
    int cur = kt & 1;
    if (kt + 1 < 8) stageW(cur ^ 1, (kt + 1) * 32);
    int k0 = kt * 32;
    union { v16h v; v8h h[2]; } fa;
    fa.h[0] = *(const v8h*)&sT[m_tile * 16 + mrow][k0 + kb];
    fa.h[1] = *(const v8h*)&sT[m_tile * 16 + mrow][k0 + kb + 16];
#pragma unroll
    for (int j = 0; j < 4; j++) {
      int nt = n_col + 4 * j;
      union { v16h v; v8h h[2]; } fb;
      fb.h[0] = *(const v8h*)&u.w[cur][nt * 16 + mrow][kb];
      fb.h[1] = *(const v8h*)&u.w[cur][nt * 16 + mrow][kb + 16];
      acc[j] = __builtin_amdgcn_wmma_f32_16x16x32_f16(
          false, fa.v, false, fb.v, (short)0, acc[j], false, false);
    }
    __syncthreads();
  }

  // write h = relu(acc + b1) into LDS (union now reused as f32 h)
#pragma unroll
  for (int j = 0; j < 4; j++) {
    int col = (n_col + 4 * j) * 16 + (lane & 15);
    int mb  = m_tile * 16 + ((lane >> 4) << 3);
    float bb = b1[col];
#pragma unroll
    for (int r = 0; r < 8; r++) {
      float v = acc[j][r] + bb;
      u.h[mb + r][col] = v > 0.f ? v : 0.f;
    }
  }
  __syncthreads();

  // logits: out[c] = h[c] . w2 + b2 (8 lanes per choice)
  {
    int c = tid >> 3, p = tid & 7;
    float s = 0.f;
    for (int k = p * 32; k < p * 32 + 32; k++) s += u.h[c][k] * w2[k];
    s += __shfl_xor(s, 1);
    s += __shfl_xor(s, 2);
    s += __shfl_xor(s, 4);
    if (p == 0) sLogit[c] = s + b2[0];
  }
  __syncthreads();

  // softmax over the 32 choices (wave 0)
  if (tid < 32) {
    float v = sLogit[tid];
    float mx = v;
#pragma unroll
    for (int m = 16; m; m >>= 1) { float o = __shfl_xor(mx, m); mx = o > mx ? o : mx; }
    float e = __expf(v - mx);
    float sum = e;
#pragma unroll
    for (int m = 16; m; m >>= 1) sum += __shfl_xor(sum, m);
    out[(size_t)node * 32 + tid] = e / sum;
  }
}

// ---------------------------------------------------------------- host driver
extern "C" void kernel_launch(void* const* d_in, const int* in_sizes, int n_in,
                              void* d_out, int out_size, void* d_ws, size_t ws_size,
                              hipStream_t stream) {
  (void)in_sizes; (void)n_in; (void)out_size; (void)ws_size;

  const float* x   = (const float*)d_in[0];
  const float* spc = (const float*)d_in[1];
  const float* rdc = (const float*)d_in[2];
  const float* fuc = (const float*)d_in[3];
  const float* roc = (const float*)d_in[4];
  const float* unc = (const float*)d_in[5];
  const float* P   = (const float*)d_in[6];
  const int* ntiD  = (const int*)d_in[7];
  const int* ei    = (const int*)d_in[8];
  const int* etiD  = (const int*)d_in[9];
  float* out = (float*)d_out;

  const int NTI_h[5] = {0, 10000, 20000, 30000, 30000};

  // params layout: pytree-flatten order (alphabetical dict keys)
  auto mlp_sz = [](int lc) { return 256 + 1 + 128 + lc * 128 + 128 + 64 * 128 + 256 * 256 + 256; };
  auto gat_sz = [](int inc, int outc, int h) { return 5 * h * 2 * outc + 4 * h * outc + 4 * inc * h * outc; };
  size_t off = 0;
  const float* FUSE = P + off; off += mlp_sz(3);
  const float* PL1  = P + off; off += gat_sz(128, 64, 1);
  const float* PL2  = P + off; off += gat_sz(64, 64, 1);
  const float* PL3  = P + off; off += gat_sz(64, 128, 2);
  const float* PL4  = P + off; off += gat_sz(256, 64, 1);
  const float* RED  = P + off; off += mlp_sz(3);
  const float* REO  = P + off; off += mlp_sz(1);
  const float* SPA  = P + off; off += mlp_sz(4);
  const float* UNR  = P + off; off += mlp_sz(2);
  auto w1_of = [](const float* blk, int lc) { return blk + 8705 + lc * 128; };

  // workspace carving (~88 MB f32 + ~1 MB f16 weights)
  float* w = (float*)d_ws;
  float* bufL    = w; w += (size_t)NNODES * 256;   // per-layer lin (xh)
  float* bufG    = w; w += (size_t)NNODES * 256;   // per-layer aggr / o1 / o3
  float* d_o2    = w; w += (size_t)NNODES * 64;
  float* d_act4  = w; w += (size_t)NNODES * 64;
  float* d_si    = w; w += (size_t)NNODES * 5 * 2;
  float* d_sj    = w; w += (size_t)NNODES * 5 * 2;
  float* d_alpha = w; w += (size_t)NEDGES * 2;
  float* d_ex    = w; w += (size_t)NEDGES * 2;
  float* d_amax  = w; w += (size_t)NNODES * 2;
  float* d_den   = w; w += (size_t)NNODES * 2;
  float* d_mean  = w; w += 128;
  _Float16* w1h  = (_Float16*)w;           // 5 * 65536 (heads' w1, transposed)
  _Float16* nw1h = w1h + 5 * 65536;        // 4*128*64
  _Float16* nw2h = nw1h + 32768;           // 4*64*64
  _Float16* nw3h = nw2h + 16384;           // 4*64*256
  _Float16* nw4h = nw3h + 65536;           // 4*256*64

  auto cdiv = [](long long a, long long b) { return (unsigned)((a + b - 1) / b); };

  // ---- one-time weight prep: f32 -> f16, transposed to [N][K] ----
  {
    const float* srcs[5] = {w1_of(SPA, 4), w1_of(RED, 3), w1_of(FUSE, 3), w1_of(REO, 1), w1_of(UNR, 2)};
    for (int i = 0; i < 5; i++)
      convert_tr_kernel<<<cdiv(65536, 256), 256, 0, stream>>>(srcs[i], w1h + (size_t)i * 65536, 256, 256, 1);
    struct { const float* gp; int inc, heads, outc; _Float16* dst; } L[4] = {
        {PL1, 128, 1, 64, nw1h}, {PL2, 64, 1, 64, nw2h},
        {PL3, 64, 2, 128, nw3h}, {PL4, 256, 1, 64, nw4h}};
    for (int i = 0; i < 4; i++) {
      int hc = L[i].heads * L[i].outc;
      const float* nw = L[i].gp + (size_t)5 * L[i].heads * 2 * L[i].outc + (size_t)4 * hc;
      convert_tr_kernel<<<cdiv((long long)4 * L[i].inc * hc, 256), 256, 0, stream>>>(
          nw, L[i].dst, L[i].inc, hc, 4);
    }
  }

  auto run_gat = [&](const float* in, int lda_in, int relu_in, const float* gp,
                     const _Float16* nwh, int inc, int heads, int outc,
                     const float* resid, float* out_o, int relu_out_flag) {
    int hc = heads * outc;
    const float* ew = gp;
    const float* nb = gp + (size_t)5 * heads * 2 * outc;
    int h_sh  = (heads == 2) ? 1 : 0;
    int oc_sh = (outc == 128) ? 7 : 6;
    // 1. lin = relu(in @ nw[t]) per node type (f16 WMMA, pre-transposed B)
    for (int t = 0; t < 4; t++) {
      int M = NTI_h[t + 1] - NTI_h[t];
      if (M <= 0) continue;
      dim3 g(cdiv(M, BM), cdiv(hc, BN));
      gemm_wmma_kernel<<<g, 256, 0, stream>>>(
          in + (size_t)NTI_h[t] * lda_in, nwh + (size_t)t * inc * hc,
          bufL + (size_t)NTI_h[t] * hc, M, inc, lda_in, hc, relu_in, 1);
    }
    // 2. attention projections (wave per (n,t,h))
    sisj_kernel<<<cdiv((long long)NNODES * 5 * heads * 32, 256), 256, 0, stream>>>(
        bufL, ew, d_si, d_sj, NNODES, heads, outc);
    // 3. init segment buffers
    fill_kernel<<<cdiv((long long)NNODES * heads, 256), 256, 0, stream>>>(d_amax, -__builtin_inff(), (long long)NNODES * heads);
    fill_kernel<<<cdiv((long long)NNODES * heads, 256), 256, 0, stream>>>(d_den, 0.f, (long long)NNODES * heads);
    fill_kernel<<<cdiv((long long)NNODES * hc, 256), 256, 0, stream>>>(bufG, 0.f, (long long)NNODES * hc);
    // 4. edge softmax + aggregation
    long long eh = (long long)NEDGES * heads;
    edge_pass1_kernel<<<cdiv(eh, 256), 256, 0, stream>>>(ei, etiD, d_si, d_sj, d_alpha, d_amax, NEDGES, h_sh);
    edge_pass2_kernel<<<cdiv(eh, 256), 256, 0, stream>>>(ei, d_alpha, d_amax, d_ex, d_den, NEDGES, h_sh);
    long long t3 = (long long)NEDGES * heads * outc;
    edge_pass3_kernel<<<cdiv(t3, 256), 256, 0, stream>>>(ei, d_ex, d_den, bufL, bufG, NEDGES, h_sh, oc_sh);
    // 5. bias (+ residual, + relu)
    bias_out_kernel<<<cdiv((long long)NNODES * hc, 256), 256, 0, stream>>>(
        bufG, nb, ntiD, resid, out_o, NNODES, hc, relu_out_flag);
  };

  // GAT stack (o1 and o3 held in bufG in place; relu folded into next GEMM's A load)
  run_gat(x,    128, 0, PL1, nw1h, 128, 1, 64,  nullptr, bufG,   0);  // o1
  run_gat(bufG, 64,  1, PL2, nw2h, 64,  1, 64,  nullptr, d_o2,   0);  // o2
  run_gat(d_o2, 64,  1, PL3, nw3h, 64,  2, 128, nullptr, bufG,   0);  // o3
  run_gat(bufG, 256, 1, PL4, nw4h, 256, 1, 64,  d_o2,    d_act4, 1);  // act4 = relu(o4+o2)

  // means for shared heads
  col_mean_kernel<<<64, 256, 0, stream>>>(d_act4, 10000, 20000, 64, d_mean);       // fuse_act
  col_mean_kernel<<<64, 256, 0, stream>>>(d_act4, 10000, 30000, 64, d_mean + 64);  // reorder_act

  // fused MLP heads + softmax
  mlp_head_kernel<<<10000, 256, 0, stream>>>(spc, d_act4 + (size_t)10000 * 64, 64, 4, SPA, w1h + 0 * 65536, out + 0);
  mlp_head_kernel<<<10000, 256, 0, stream>>>(rdc, d_act4 + (size_t)20000 * 64, 64, 3, RED, w1h + 1 * 65536, out + 320000);
  mlp_head_kernel<<<4, 256, 0, stream>>>(fuc, d_mean,      0, 3, FUSE, w1h + 2 * 65536, out + 640000);
  mlp_head_kernel<<<4, 256, 0, stream>>>(roc, d_mean + 64, 0, 1, REO,  w1h + 3 * 65536, out + 640128);
  mlp_head_kernel<<<4, 256, 0, stream>>>(unc, d_mean + 64, 0, 2, UNR,  w1h + 4 * 65536, out + 640256);
}
